// GNNPolicyNetwork_16355235463220
// MI455X (gfx1250) — matrix operand order repaired and checked by reading, not compiled, not run on hardware
//
#include <hip/hip_runtime.h>
#include <hip/hip_bf16.h>
#include <stdint.h>

typedef __bf16 v16bf __attribute__((ext_vector_type(16)));
typedef float  v8f   __attribute__((ext_vector_type(8)));

#define DEVFN static __device__ __forceinline__

constexpr int Bc  = 4;
constexpr int Nn  = 10000;
constexpr int Ec  = 320000;
constexpr int EAc = 80000;
constexpr int ND  = 64;
constexpr int ED  = 16;
constexpr int HD  = 128;

union Frag16 { v16bf v; uint32_t d[8]; unsigned short u[16]; };

DEVFN unsigned short f2bf(float f) {
  union { float f; uint32_t u; } c; c.f = f;
  uint32_t r = c.u + 0x7FFFu + ((c.u >> 16) & 1u);   // round-to-nearest-even
  return (unsigned short)(r >> 16);
}

// Guaranteed native f32 scatter-add (no CAS loop). Tracked by STOREcnt;
// implicit S_WAIT_IDLE at S_ENDPGM covers completion.
DEVFN void atomic_add_f32(float* p, float v) {
  asm volatile("global_atomic_add_f32 %0, %1, off" :: "v"(p), "v"(v) : "memory");
}

// ---- vectorized gather helpers (4 elements per op) ----
DEVFN void gather4_f32(unsigned short* dst, const float* __restrict__ src) {
  float4 f = *(const float4*)src;                 // global_load_b128
  uint2 u;
  u.x = (uint32_t)f2bf(f.x) | ((uint32_t)f2bf(f.y) << 16);
  u.y = (uint32_t)f2bf(f.z) | ((uint32_t)f2bf(f.w) << 16);
  *(uint2*)dst = u;                               // ds_store_b64
}
DEVFN void copy4_bf(unsigned short* dst, const unsigned short* __restrict__ src) {
  *(uint2*)dst = *(const uint2*)src;              // global_load_b64 + ds_store_b64
}
DEVFN void zero4(unsigned short* dst) {
  uint2 z; z.x = 0u; z.y = 0u;
  *(uint2*)dst = z;
}

// ---- A fragment from LDS: 16-bit A 16x32 layout (ISA 7.12.2) ----
// lanes 0-15: M=lane, K = klo..klo+7 (v0-3) and klo+16..klo+23 (v4-7), klo=ks*32
// lanes 16-31: same M, klo = ks*32+8
DEVFN v16bf ld_a_lds(const unsigned short* X, int stride, int mrowBase, int kstep) {
  const int lane = threadIdx.x & 31;
  const int row  = mrowBase + (lane & 15);
  const int klo  = kstep * 32 + ((lane & 16) ? 8 : 0);
  const uint32_t* p = (const uint32_t*)(X + row * stride + klo);
  const uint32_t* q = (const uint32_t*)(X + row * stride + klo + 16);
  Frag16 f;
#pragma unroll
  for (int i = 0; i < 4; ++i) { f.d[i] = p[i]; f.d[4 + i] = q[i]; }
  return f.v;
}

// ---- B fragment from pre-packed global weights: contiguous 32B per lane ----
DEVFN v16bf ld_b_glb(const unsigned short* Wp, int NB, int kstep, int nb) {
  const int lane = threadIdx.x & 31;
  const uint32_t* p = (const uint32_t*)(Wp + (size_t)(((kstep * NB + nb) * 32 + lane) * 16));
  Frag16 f;
#pragma unroll
  for (int i = 0; i < 8; ++i) f.d[i] = p[i];
  return f.v;
}

// 16xKx(4*16) per-wave GEMM: 4 adjacent 16x16 N-tiles, one 16-row M block.
template<int KSTEPS>
DEVFN void gemm4(const unsigned short* X, int stride, int mrowBase,
                 const unsigned short* Wp, int NB, int nb0, v8f acc[4]) {
#pragma unroll
  for (int ks = 0; ks < KSTEPS; ++ks) {
    v16bf a = ld_a_lds(X, stride, mrowBase, ks);
#pragma unroll
    for (int t = 0; t < 4; ++t) {
      v16bf bm = ld_b_glb(Wp, NB, ks, nb0 + t);
      acc[t] = __builtin_amdgcn_wmma_f32_16x16x32_bf16(false, a, false, bm,
                                                       (short)0, acc[t], false, false);
    }
  }
}

// ---- weight packer: fp32 (K x Nw) -> bf16 WMMA-B fragment layout ----
__global__ void pack_weights(const float* __restrict__ src, unsigned short* __restrict__ dst,
                             int K, int Nw, int KB, int NB) {
  int idx = blockIdx.x * blockDim.x + threadIdx.x;
  int total = KB * NB * 512;
  if (idx >= total) return;
  int i    = idx & 15;
  int lane = (idx >> 4) & 31;
  int nb   = (idx >> 9) % NB;
  int kb   = (idx >> 9) / NB;
  int k = kb * 32 + ((lane >> 4) << 3) + (i & 7) + ((i >> 3) << 4);   // lanes 16-31 hold K+8 (table order)
  int n = nb * 16 + (lane & 15);
  float v = (k < K && n < Nw) ? src[k * Nw + n] : 0.0f;
  dst[idx] = f2bf(v);
}

// ---- edge message kernel: gather -> MLP(2 GEMMs) -> scatter-add ----
template<int LAYER, int KP, int KSTEPS>
__global__ void __launch_bounds__(256) msg_kernel(
    const float* __restrict__ nodes_f32,       // layer 1 node source (fp32, 64)
    const unsigned short* __restrict__ h_bf16, // layer 2 node source (bf16, 128)
    const int* __restrict__ eidx, const float* __restrict__ eattr,
    const unsigned short* __restrict__ Wp1, const float* __restrict__ b1,
    const unsigned short* __restrict__ Wp2, const float* __restrict__ b2,
    float* __restrict__ agg) {
  __shared__ unsigned short X[64 * KP];
  __shared__ unsigned short Y[64 * HD];
  __shared__ int rIdx[64];

  const int b    = blockIdx.y;
  const int row  = threadIdx.x >> 2;
  const int part = threadIdx.x & 3;
  const int e    = blockIdx.x * 64 + row;      // E divisible by 64
  const int r    = eidx[e];
  const int s    = eidx[Ec + e];
  if (part == 0) rIdx[row] = r;

  constexpr int IN  = (LAYER == 1) ? ND : HD;
  constexpr int CPT = KP / 4;                  // columns per thread
#pragma unroll
  for (int jj = 0; jj < CPT; jj += 4) {        // all segment bounds are mult of 4
    int j = part * CPT + jj;
    unsigned short* dst = &X[row * KP + j];
    if (LAYER == 1) {
      if (j < IN)               gather4_f32(dst, &nodes_f32[((size_t)b * Nn + r) * ND + j]);
      else if (j < 2 * IN)      gather4_f32(dst, &nodes_f32[((size_t)b * Nn + s) * ND + (j - IN)]);
      else if (j < 2 * IN + ED) gather4_f32(dst, &eattr[((size_t)b * Ec + e) * ED + (j - 2 * IN)]);
      else                      zero4(dst);
    } else {
      if (j < IN)               copy4_bf(dst, &h_bf16[((size_t)b * Nn + r) * HD + j]);
      else if (j < 2 * IN)      copy4_bf(dst, &h_bf16[((size_t)b * Nn + s) * HD + (j - IN)]);
      else if (j < 2 * IN + ED) gather4_f32(dst, &eattr[((size_t)b * Ec + e) * ED + (j - 2 * IN)]);
      else                      zero4(dst);
    }
  }
  __syncthreads();

  const int wid  = threadIdx.x >> 5;
  const int lane = threadIdx.x & 31;
  const int mb   = (wid & 3) * 16;
  const int nb0  = (wid >> 2) * 4;

  { // GEMM1 + bias + relu -> Y (bf16)
    v8f acc[4] = {};
    gemm4<KSTEPS>(X, KP, mb, Wp1, 8, nb0, acc);
#pragma unroll
    for (int t = 0; t < 4; ++t) {
      int col = (nb0 + t) * 16 + (lane & 15);
      int rr  = mb + ((lane & 16) ? 8 : 0);
      float bias = b1[col];
#pragma unroll
      for (int v = 0; v < 8; ++v)
        Y[(rr + v) * HD + col] = f2bf(fmaxf(acc[t][v] + bias, 0.0f));
    }
  }
  __syncthreads();

  { // GEMM2 + bias -> scatter-add into agg[b, r, :]
    v8f acc[4] = {};
    gemm4<4>(Y, HD, mb, Wp2, 8, nb0, acc);
#pragma unroll
    for (int t = 0; t < 4; ++t) {
      int col = (nb0 + t) * 16 + (lane & 15);
      int rr  = mb + ((lane & 16) ? 8 : 0);
      float bias = b2[col];
#pragma unroll
      for (int v = 0; v < 8; ++v) {
        int tgt = rIdx[rr + v];
        atomic_add_f32(&agg[((size_t)b * Nn + tgt) * HD + col], acc[t][v] + bias);
      }
    }
  }
}

// ---- node update kernel: relu([x, agg] @ uw + ub) -> h (bf16) ----
template<int LAYER, int KP, int KSTEPS>
__global__ void __launch_bounds__(256) upd_kernel(
    const float* __restrict__ nodes_f32,
    const unsigned short* __restrict__ h_in,
    const float* __restrict__ agg,
    const unsigned short* __restrict__ Wp, const float* __restrict__ bias,
    unsigned short* __restrict__ h_out) {
  __shared__ unsigned short X[64 * KP];
  const int b    = blockIdx.y;
  const int row  = threadIdx.x >> 2;
  const int part = threadIdx.x & 3;
  const int n    = blockIdx.x * 64 + row;
  const bool valid = (n < Nn);
  constexpr int IN  = (LAYER == 1) ? ND : HD;    // KP == IN + HD exactly
  constexpr int CPT = KP / 4;
#pragma unroll
  for (int jj = 0; jj < CPT; jj += 4) {
    int j = part * CPT + jj;
    unsigned short* dst = &X[row * KP + j];
    if (!valid) {
      zero4(dst);
    } else if (j < IN) {
      if (LAYER == 1) gather4_f32(dst, &nodes_f32[((size_t)b * Nn + n) * ND + j]);
      else            copy4_bf(dst, &h_in[((size_t)b * Nn + n) * HD + j]);
    } else {
      gather4_f32(dst, &agg[((size_t)b * Nn + n) * HD + (j - IN)]);
    }
  }
  __syncthreads();

  const int wid  = threadIdx.x >> 5;
  const int lane = threadIdx.x & 31;
  const int mb   = (wid & 3) * 16;
  const int nb0  = (wid >> 2) * 4;
  v8f acc[4] = {};
  gemm4<KSTEPS>(X, KP, mb, Wp, 8, nb0, acc);
#pragma unroll
  for (int t = 0; t < 4; ++t) {
    int col = (nb0 + t) * 16 + (lane & 15);
    int rr  = mb + ((lane & 16) ? 8 : 0);
    float bv = bias[col];
#pragma unroll
    for (int v = 0; v < 8; ++v) {
      int nn = blockIdx.x * 64 + rr + v;
      if (nn < Nn)
        h_out[((size_t)b * Nn + nn) * HD + col] = f2bf(fmaxf(acc[t][v] + bv, 0.0f));
    }
  }
}

// ---- action head: masked gather -> MLP(272->128->2) -> pair-mean -> mean/std ----
__global__ void __launch_bounds__(256) head_kernel(
    const unsigned short* __restrict__ h2,
    const int* __restrict__ eidx, const float* __restrict__ eattr,
    const int* __restrict__ mask,
    const unsigned short* __restrict__ Wp1, const float* __restrict__ b1,
    const unsigned short* __restrict__ Wp2, const float* __restrict__ b2,
    float* __restrict__ out) {
  constexpr int KP = 288;
  __shared__ unsigned short X[64 * KP];
  __shared__ unsigned short Y[64 * HD];
  const int b    = blockIdx.y;
  const int row  = threadIdx.x >> 2;
  const int part = threadIdx.x & 3;
  const int ea   = blockIdx.x * 64 + row;      // EA divisible by 64
  const int a    = mask[ea];
  const int r    = eidx[a];
  const int s    = eidx[Ec + a];
  constexpr int CPT = KP / 4;                  // 72
#pragma unroll
  for (int jj = 0; jj < CPT; jj += 4) {
    int j = part * CPT + jj;
    unsigned short* dst = &X[row * KP + j];
    if (j < HD)               copy4_bf(dst, &h2[((size_t)b * Nn + r) * HD + j]);
    else if (j < 2 * HD)      copy4_bf(dst, &h2[((size_t)b * Nn + s) * HD + (j - HD)]);
    else if (j < 2 * HD + ED) gather4_f32(dst, &eattr[((size_t)b * Ec + a) * ED + (j - 2 * HD)]);
    else                      zero4(dst);
  }
  __syncthreads();

  const int wid  = threadIdx.x >> 5;
  const int lane = threadIdx.x & 31;
  const int mb   = (wid & 3) * 16;
  const int nb0  = (wid >> 2) * 4;
  { // GEMM1 272->128, relu
    v8f acc[4] = {};
    gemm4<9>(X, KP, mb, Wp1, 8, nb0, acc);
#pragma unroll
    for (int t = 0; t < 4; ++t) {
      int col = (nb0 + t) * 16 + (lane & 15);
      int rr  = mb + ((lane & 16) ? 8 : 0);
      float bias = b1[col];
#pragma unroll
      for (int v = 0; v < 8; ++v)
        Y[(rr + v) * HD + col] = f2bf(fmaxf(acc[t][v] + bias, 0.0f));
    }
  }
  __syncthreads();
  if (wid < 4) { // GEMM2 128->2 (N padded to 16 with zero cols), one N-tile
    v8f acc = {};
#pragma unroll
    for (int ks = 0; ks < 4; ++ks) {
      v16bf av = ld_a_lds(Y, HD, mb, ks);
      v16bf bv = ld_b_glb(Wp2, 1, ks, 0);
      acc = __builtin_amdgcn_wmma_f32_16x16x32_bf16(false, av, false, bv,
                                                    (short)0, acc, false, false);
    }
    int col = lane & 15;
    if (col < 2) {
      int rbase = mb + ((lane & 16) ? 8 : 0);
      float bias = b2[col];
      constexpr int EAh = EAc / 2;
#pragma unroll
      for (int v = 0; v < 8; v += 2) {
        int rowA = rbase + v;                            // even row of the pair
        float m  = 0.5f * (acc[v] + acc[v + 1]) + bias;  // pair mean (bias invariant)
        int j = (blockIdx.x * 64 + rowA) >> 1;
        if (col == 0) {
          out[(size_t)b * EAh + j] = m;
        } else {
          float ls = fminf(fmaxf(m, -20.0f), 2.0f);
          out[(size_t)Bc * EAh + (size_t)b * EAh + j] = expf(ls);
        }
      }
    }
  }
}

extern "C" void kernel_launch(void* const* d_in, const int* in_sizes, int n_in,
                              void* d_out, int out_size, void* d_ws, size_t ws_size,
                              hipStream_t stream) {
  (void)in_sizes; (void)n_in; (void)out_size; (void)ws_size;
  const float* nodes  = (const float*)d_in[0];
  const int*   eidx   = (const int*)d_in[1];
  const float* eattr  = (const float*)d_in[2];
  const int*   mask   = (const int*)d_in[3];
  const float* g1_mw1 = (const float*)d_in[4];
  const float* g1_mb1 = (const float*)d_in[5];
  const float* g1_mw2 = (const float*)d_in[6];
  const float* g1_mb2 = (const float*)d_in[7];
  const float* g1_uw  = (const float*)d_in[8];
  const float* g1_ub  = (const float*)d_in[9];
  const float* g2_mw1 = (const float*)d_in[10];
  const float* g2_mb1 = (const float*)d_in[11];
  const float* g2_mw2 = (const float*)d_in[12];
  const float* g2_mb2 = (const float*)d_in[13];
  const float* g2_uw  = (const float*)d_in[14];
  const float* g2_ub  = (const float*)d_in[15];
  const float* a_w1   = (const float*)d_in[16];
  const float* a_b1   = (const float*)d_in[17];
  const float* a_w2   = (const float*)d_in[18];
  const float* a_b2   = (const float*)d_in[19];

  char* ws = (char*)d_ws;
  auto alloc = [&](size_t bytes) {
    char* p = ws;
    ws += (bytes + 255) & ~(size_t)255;
    return p;
  };
  auto packed_bytes = [](int K, int Nw) {
    return (size_t)((K + 31) / 32) * ((Nw + 15) / 16) * 512 * 2;
  };

  unsigned short* wp_g1_mw1 = (unsigned short*)alloc(packed_bytes(144, 128));
  unsigned short* wp_g1_mw2 = (unsigned short*)alloc(packed_bytes(128, 128));
  unsigned short* wp_g1_uw  = (unsigned short*)alloc(packed_bytes(192, 128));
  unsigned short* wp_g2_mw1 = (unsigned short*)alloc(packed_bytes(272, 128));
  unsigned short* wp_g2_mw2 = (unsigned short*)alloc(packed_bytes(128, 128));
  unsigned short* wp_g2_uw  = (unsigned short*)alloc(packed_bytes(256, 128));
  unsigned short* wp_a_w1   = (unsigned short*)alloc(packed_bytes(272, 128));
  unsigned short* wp_a_w2   = (unsigned short*)alloc(packed_bytes(128, 2));
  unsigned short* h1  = (unsigned short*)alloc((size_t)Bc * Nn * HD * 2);
  unsigned short* h2  = (unsigned short*)alloc((size_t)Bc * Nn * HD * 2);
  float* agg1 = (float*)alloc((size_t)Bc * Nn * HD * 4);
  float* agg2 = (float*)alloc((size_t)Bc * Nn * HD * 4);

  auto pack = [&](const float* src, unsigned short* dst, int K, int Nw) {
    int KB = (K + 31) / 32, NB = (Nw + 15) / 16;
    int total = KB * NB * 512;
    pack_weights<<<(total + 255) / 256, 256, 0, stream>>>(src, dst, K, Nw, KB, NB);
  };
  pack(g1_mw1, wp_g1_mw1, 144, 128);
  pack(g1_mw2, wp_g1_mw2, 128, 128);
  pack(g1_uw,  wp_g1_uw,  192, 128);
  pack(g2_mw1, wp_g2_mw1, 272, 128);
  pack(g2_mw2, wp_g2_mw2, 128, 128);
  pack(g2_uw,  wp_g2_uw,  256, 128);
  pack(a_w1,   wp_a_w1,   272, 128);
  pack(a_w2,   wp_a_w2,   128, 2);

  hipMemsetAsync(agg1, 0, (size_t)Bc * Nn * HD * 4, stream);
  hipMemsetAsync(agg2, 0, (size_t)Bc * Nn * HD * 4, stream);

  const dim3 blk(256);
  // Layer 1
  msg_kernel<1, 160, 5><<<dim3(Ec / 64, Bc), blk, 0, stream>>>(
      nodes, nullptr, eidx, eattr, wp_g1_mw1, g1_mb1, wp_g1_mw2, g1_mb2, agg1);
  upd_kernel<1, 192, 6><<<dim3((Nn + 63) / 64, Bc), blk, 0, stream>>>(
      nodes, nullptr, agg1, wp_g1_uw, g1_ub, h1);
  // Layer 2
  msg_kernel<2, 288, 9><<<dim3(Ec / 64, Bc), blk, 0, stream>>>(
      nullptr, h1, eidx, eattr, wp_g2_mw1, g2_mb1, wp_g2_mw2, g2_mb2, agg2);
  upd_kernel<2, 256, 8><<<dim3((Nn + 63) / 64, Bc), blk, 0, stream>>>(
      nullptr, h1, agg2, wp_g2_uw, g2_ub, h2);
  // Action head
  head_kernel<<<dim3(EAc / 64, Bc), blk, 0, stream>>>(
      h2, eidx, eattr, mask, wp_a_w1, a_b1, wp_a_w2, a_b2, (float*)d_out);
}
